// AgentDecoder_84859963834502
// MI455X (gfx1250) — matrix operand order repaired and checked by reading, not compiled, not run on hardware
//
#include <hip/hip_runtime.h>
#include <hip/hip_bf16.h>
#include <math.h>

#define BB 128
#define AA 64
#define DD 256
#define TT 80
#define MM 6
#define BA (BB*AA)            // 8192
#define ROWS_CTX (BA*4)       // 32768
#define ROWS_MODE (BA*MM)     // 49152
#define TRAJ_ELEMS (BB*MM*AA*TT*3)  // 11796480

typedef __attribute__((ext_vector_type(16))) _Float16 v16h;
typedef __attribute__((ext_vector_type(8)))  _Float16 v8h;
typedef __attribute__((ext_vector_type(8)))  float    v8f;

// ---------------------------------------------------------------------------
// WMMA fragment loaders (wave32, V_WMMA_F32_16X16X32_F16 layouts from ISA 7.12.2)
// A 16x32 f16: lane m (0..15) holds row m; lanes<16: K = k0+{0..7, 16..23},
//              lanes>=16: K = k0+{8..15, 24..31}; 2 halves per VGPR.
// B 32x16 f16 (stored N-major [n][k]): lane n (0..15) col n; lanes<16 K=k0+{0..7, 16..23},
//              lanes>=16 K=k0+{8..15, 24..31} (same order as A).
// C/D f32: c[r] = D[m0 + r + (lane>>4)*8][n0 + (lane&15)]
// ---------------------------------------------------------------------------
__device__ __forceinline__ v16h make_frag(v8h lo, v8h hi) {
  v16h r;
#pragma unroll
  for (int i = 0; i < 8; ++i) { r[i] = lo[i]; r[i + 8] = hi[i]; }
  return r;
}

__device__ __forceinline__ v16h load_a_row(const _Float16* __restrict__ prow, int k0) {
  int lane = threadIdx.x & 31;
  int kb = k0 + ((lane >> 4) << 3);
  v8h lo = *(const v8h*)(prow + kb);
  v8h hi = *(const v8h*)(prow + kb + 16);
  return make_frag(lo, hi);
}

__device__ __forceinline__ v16h load_b_col(const _Float16* __restrict__ Bt, int ldb,
                                           int n0, int k0) {
  int lane = threadIdx.x & 31;
  const _Float16* p = Bt + (long)(n0 + (lane & 15)) * ldb + k0 + ((lane >> 4) << 3);
  v8h lo = *(const v8h*)(p);
  v8h hi = *(const v8h*)(p + 16);
  return make_frag(lo, hi);
}

#define WMMA_F16(A_, B_, C_) \
  __builtin_amdgcn_wmma_f32_16x16x32_f16(false, (A_), false, (B_), (short)0, (C_), false, false)

// One 32xK-step of the 32x64 tile: 2 A frags, 4 B frags, 8 WMMAs.
__device__ __forceinline__ void tile_step(const _Float16* __restrict__ p0, int ka0,
                                          const _Float16* __restrict__ p1, int ka1,
                                          const _Float16* __restrict__ Bt, int ldb,
                                          int n0, int kb, v8f acc[8]) {
  v16h a0 = load_a_row(p0, ka0);
  v16h a1 = load_a_row(p1, ka1);
  v16h b0 = load_b_col(Bt, ldb, n0,      kb);
  v16h b1 = load_b_col(Bt, ldb, n0 + 16, kb);
  v16h b2 = load_b_col(Bt, ldb, n0 + 32, kb);
  v16h b3 = load_b_col(Bt, ldb, n0 + 48, kb);
  acc[0] = WMMA_F16(a0, b0, acc[0]);
  acc[1] = WMMA_F16(a0, b1, acc[1]);
  acc[2] = WMMA_F16(a0, b2, acc[2]);
  acc[3] = WMMA_F16(a0, b3, acc[3]);
  acc[4] = WMMA_F16(a1, b0, acc[4]);
  acc[5] = WMMA_F16(a1, b1, acc[5]);
  acc[6] = WMMA_F16(a1, b2, acc[6]);
  acc[7] = WMMA_F16(a1, b3, acc[7]);
}

// ---------------------------------------------------------------------------
// Generic WMMA GEMM: each wave computes a 32(M) x 64(N) tile; K in steps of 32.
// EPI 0: f16 store of acc+bias          EPI 1: f16 store of ELU(acc+bias)
// EPI 2: f32 store acc+bias+mode_queries[r%6] (pre-LN residual)
// EPI 3: th2: trajs (transposed, +current_state) to f32 out + f16 flat copy
// ---------------------------------------------------------------------------
template <int EPI>
__launch_bounds__(256)
__global__ void gemm_wmma(const _Float16* __restrict__ A, int lda,
                          const _Float16* __restrict__ Bt, int ldb,
                          const float* __restrict__ bias,
                          int Mrows, int Ncols, int K,
                          _Float16* __restrict__ outH, int ldoH,
                          float* __restrict__ outF,
                          const float* __restrict__ extra0) {
  int nt  = (Ncols + 63) >> 6;
  int wid = blockIdx.x * 8 + (threadIdx.x >> 5);
  int mt  = wid / nt;
  int n0  = (wid - mt * nt) << 6;
  int m0  = mt << 5;                              // 32 rows per wave
  if (m0 >= Mrows) return;                        // wave-uniform
  int lane = threadIdx.x & 31;
  const _Float16* prow0 = A + (long)(m0 + (lane & 15)) * lda;
  const _Float16* prow1 = prow0 + (long)16 * lda;

  v8f acc[8] = {{}, {}, {}, {}, {}, {}, {}, {}};
  for (int k0 = 0; k0 < K; k0 += 32)
    tile_step(prow0, k0, prow1, k0, Bt, ldb, n0, k0, acc);

  int rb0   = m0 + ((lane >> 4) << 3);
  int cbase = n0 + (lane & 15);
#pragma unroll
  for (int half = 0; half < 2; ++half) {
    int rbase = rb0 + 16 * half;
#pragma unroll
    for (int j = 0; j < 4; ++j) {
      int c = cbase + 16 * j;
#pragma unroll
      for (int rr = 0; rr < 8; ++rr) {
        int r = rbase + rr;
        float v = acc[half * 4 + j][rr];
        if (EPI == 0) {
          if (c < Ncols) outH[(long)r * ldoH + c] = (_Float16)(v + bias[c]);
        } else if (EPI == 1) {
          if (c < Ncols) {
            v += bias[c];
            v = v > 0.f ? v : (__expf(v) - 1.f);
            outH[(long)r * ldoH + c] = (_Float16)v;
          }
        } else if (EPI == 2) {
          if (c < Ncols) outF[(long)r * DD + c] = v + bias[c] + extra0[(r % MM) * DD + c];
        } else {  // EPI 3
          _Float16 hv = (_Float16)0.f;
          if (c < TT * 3) {
            v += bias[c];
            hv = (_Float16)v;
            int ba = r / MM, m = r - ba * MM;
            int b = ba >> 6, a = ba & 63;
            int t = c / 3,  d = c - t * 3;
            outF[(((long)(b * MM + m) * AA + a) * TT + t) * 3 + d] = v + extra0[ba * 3 + d];
          }
          outH[(long)r * 256 + c] = hv;  // padded flat copy for te1 (pad cols zero)
        }
      }
    }
  }
}

// ---------------------------------------------------------------------------
// sh1 GEMM (K=1344, biggest FLOP block): A is the virtual concat
// [mf(256) | te(64) | gctx(1024)]. The K loop is split into three straight-line
// phases (segment boundaries are compile-time constants) so the hot loop has
// no per-step pointer-select branches. 32x64 tile per wave.
// ---------------------------------------------------------------------------
__launch_bounds__(256)
__global__ void gemm_sh1(const _Float16* __restrict__ mf,   // [49152][256]
                         const _Float16* __restrict__ te,   // [49152][64]
                         const _Float16* __restrict__ ctx,  // [32768][256]
                         const _Float16* __restrict__ Bt,   // [256][1344]
                         const float* __restrict__ bias,
                         _Float16* __restrict__ outH) {     // ELU, [49152][256]
  const int nt = 4;
  int wid = blockIdx.x * 8 + (threadIdx.x >> 5);
  int mt  = wid / nt;
  int n0  = (wid - mt * nt) << 6;
  int m0  = mt << 5;
  if (m0 >= ROWS_MODE) return;
  int lane = threadIdx.x & 31;
  int r0 = m0 + (lane & 15);
  int r1 = r0 + 16;
  const _Float16* mfrow0 = mf + (long)r0 * 256;
  const _Float16* mfrow1 = mf + (long)r1 * 256;
  const _Float16* terow0 = te + (long)r0 * 64;
  const _Float16* terow1 = te + (long)r1 * 64;
  const _Float16* ctxrow0 = ctx + (long)(r0 / MM) * 4 * 256;
  const _Float16* ctxrow1 = ctx + (long)(r1 / MM) * 4 * 256;

  v8f acc[8] = {{}, {}, {}, {}, {}, {}, {}, {}};
  // Phase 1: mode_features, K = [0, 256)
  for (int k0 = 0; k0 < 256; k0 += 32)
    tile_step(mfrow0, k0, mfrow1, k0, Bt, 1344, n0, k0, acc);
  // Phase 2: traj_embed, K = [256, 320)
#pragma unroll
  for (int k0 = 0; k0 < 64; k0 += 32)
    tile_step(terow0, k0, terow1, k0, Bt, 1344, n0, 256 + k0, acc);
  // Phase 3: gctx (4 ctx sub-rows), K = [320, 1344)
#pragma unroll
  for (int j = 0; j < 4; ++j)
    for (int k0 = 0; k0 < 256; k0 += 32)
      tile_step(ctxrow0 + j * 256, k0, ctxrow1 + j * 256, k0,
                Bt, 1344, n0, 320 + j * 256 + k0, acc);

  int rb0   = m0 + ((lane >> 4) << 3);
  int cbase = n0 + (lane & 15);
#pragma unroll
  for (int half = 0; half < 2; ++half) {
    int rbase = rb0 + 16 * half;
#pragma unroll
    for (int j = 0; j < 4; ++j) {
      int c = cbase + 16 * j;
#pragma unroll
      for (int rr = 0; rr < 8; ++rr) {
        float v = acc[half * 4 + j][rr] + bias[c];
        v = v > 0.f ? v : (__expf(v) - 1.f);
        outH[(long)(rbase + rr) * 256 + c] = (_Float16)v;
      }
    }
  }
}

// ---------------------------------------------------------------------------
// Small / elementwise kernels
// ---------------------------------------------------------------------------
__global__ void wconv(const float* __restrict__ W, _Float16* __restrict__ Wt,
                      int K, int N, int Kpad, int Npad) {
  long idx = (long)blockIdx.x * blockDim.x + threadIdx.x;
  if (idx >= (long)Npad * Kpad) return;
  int n = (int)(idx / Kpad), k = (int)(idx % Kpad);
  Wt[idx] = (n < N && k < K) ? (_Float16)W[(long)k * N + n] : (_Float16)0.f;
}

__global__ void pack_inputs(const float* __restrict__ a0, const float* __restrict__ a1,
                            const float* __restrict__ a2, const float* __restrict__ a3,
                            _Float16* __restrict__ X) {
  long idx = (long)blockIdx.x * blockDim.x + threadIdx.x;
  if (idx >= (long)ROWS_CTX * DD) return;
  long ba = idx >> 10;                 // / (4*256)
  int rem = (int)(idx & 1023);
  int j = rem >> 8, c = rem & 255;
  const float* src = j == 0 ? a0 : j == 1 ? a1 : j == 2 ? a2 : a3;
  X[idx] = (_Float16)src[ba * DD + c];
}

__global__ void qproj(const float* __restrict__ mq, const float* __restrict__ W,
                      const float* __restrict__ b, float* __restrict__ qh) {
  int m = blockIdx.x, c = threadIdx.x;
  float s = b[c];
  for (int k = 0; k < DD; ++k) s += mq[m * DD + k] * W[(long)k * DD + c];
  qh[m * DD + c] = s;
}

__launch_bounds__(256)
__global__ void attn_kernel(const float* __restrict__ qh, const _Float16* __restrict__ kf,
                            const _Float16* __restrict__ vf, _Float16* __restrict__ ao) {
  int idx = blockIdx.x * blockDim.x + threadIdx.x;   // BA*M*H
  if (idx >= BA * MM * 4) return;
  int h = idx & 3;
  int m = (idx >> 2) % MM;
  int ba = idx / (4 * MM);
  const float* q = qh + m * DD + h * 64;
  const _Float16* kb = kf + (long)ba * 4 * DD + h * 64;
  const _Float16* vb = vf + (long)ba * 4 * DD + h * 64;
  float w[4];
#pragma unroll
  for (int s = 0; s < 4; ++s) {
    float dot = 0.f;
    for (int d = 0; d < 64; ++d) dot += q[d] * (float)kb[(long)s * DD + d];
    w[s] = 1.f / (1.f + __expf(-dot * 0.125f));    // sigmoid(score/sqrt(64))
  }
  _Float16* o = ao + ((long)ba * MM + m) * DD + h * 64;
  for (int d = 0; d < 64; ++d) {
    float acc = 0.f;
#pragma unroll
    for (int s = 0; s < 4; ++s) acc += w[s] * (float)vb[(long)s * DD + d];
    o[d] = (_Float16)acc;
  }
}

__launch_bounds__(256)
__global__ void ln_rows(const float* __restrict__ x, const float* __restrict__ g,
                        const float* __restrict__ b, _Float16* __restrict__ mf) {
  int row = blockIdx.x * 8 + (threadIdx.x >> 5);
  if (row >= ROWS_MODE) return;
  int lane = threadIdx.x & 31;
  const float* xr = x + (long)row * DD;
  float v[8], s = 0.f, s2 = 0.f;
#pragma unroll
  for (int i = 0; i < 8; ++i) { v[i] = xr[i * 32 + lane]; s += v[i]; s2 += v[i] * v[i]; }
#pragma unroll
  for (int d = 16; d >= 1; d >>= 1) { s += __shfl_xor(s, d, 32); s2 += __shfl_xor(s2, d, 32); }
  float mu = s * (1.f / 256.f);
  float rs = rsqrtf(s2 * (1.f / 256.f) - mu * mu + 1e-5f);
  _Float16* o = mf + (long)row * DD;
#pragma unroll
  for (int i = 0; i < 8; ++i) {
    int c = i * 32 + lane;
    o[c] = (_Float16)((v[i] - mu) * rs * g[c] + b[c]);
  }
}

__launch_bounds__(256)
__global__ void score_out(const _Float16* __restrict__ gbuf, const float* __restrict__ w,
                          const float* __restrict__ b, float* __restrict__ out) {
  int row = blockIdx.x * 8 + (threadIdx.x >> 5);
  if (row >= ROWS_MODE) return;
  int lane = threadIdx.x & 31;
  const _Float16* gr = gbuf + (long)row * 256;
  float s = 0.f;
#pragma unroll
  for (int i = 0; i < 8; ++i) { int c = i * 32 + lane; s += (float)gr[c] * w[c]; }
#pragma unroll
  for (int d = 16; d >= 1; d >>= 1) s += __shfl_xor(s, d, 32);
  if (lane == 0) {
    int ba = row / MM, m = row - ba * MM;
    int bb = ba >> 6, a = ba & 63;
    out[(bb * MM + m) * AA + a] = s + b[0];
  }
}

// ---------------------------------------------------------------------------
extern "C" void kernel_launch(void* const* d_in, const int* in_sizes, int n_in,
                              void* d_out, int out_size, void* d_ws, size_t ws_size,
                              hipStream_t stream) {
  const float* agent_map   = (const float*)d_in[0];
  const float* agent_agent = (const float*)d_in[1];
  const float* neigh       = (const float*)d_in[2];
  const float* cur_state   = (const float*)d_in[3];
  const float* map_feat    = (const float*)d_in[4];
  const float* mode_q = (const float*)d_in[5];
  const float* ctx_W = (const float*)d_in[6],  *ctx_b = (const float*)d_in[7];
  const float* q_W   = (const float*)d_in[8],  *q_b   = (const float*)d_in[9];
  const float* k_W   = (const float*)d_in[10], *k_b   = (const float*)d_in[11];
  const float* v_W   = (const float*)d_in[12], *v_b   = (const float*)d_in[13];
  const float* o_W   = (const float*)d_in[14], *o_b   = (const float*)d_in[15];
  const float* ln_g  = (const float*)d_in[16], *ln_b  = (const float*)d_in[17];
  const float* th1_W = (const float*)d_in[18], *th1_b = (const float*)d_in[19];
  const float* th2_W = (const float*)d_in[20], *th2_b = (const float*)d_in[21];
  const float* te1_W = (const float*)d_in[22], *te1_b = (const float*)d_in[23];
  const float* te2_W = (const float*)d_in[24], *te2_b = (const float*)d_in[25];
  const float* sh1_W = (const float*)d_in[26], *sh1_b = (const float*)d_in[27];
  const float* sh2_W = (const float*)d_in[28], *sh2_b = (const float*)d_in[29];

  float* trajs_out  = (float*)d_out;
  float* scores_ptr = (float*)d_out + TRAJ_ELEMS;

  // Workspace bump allocator (256B aligned)
  char* ws = (char*)d_ws;
  size_t off = 0;
  auto alloc = [&](size_t bytes) -> void* {
    void* p = ws + off;
    off = (off + bytes + 255) & ~(size_t)255;
    return p;
  };
  _Float16* Wt_ctx = (_Float16*)alloc(256 * 256 * 2);
  _Float16* Wt_k   = (_Float16*)alloc(256 * 256 * 2);
  _Float16* Wt_v   = (_Float16*)alloc(256 * 256 * 2);
  _Float16* Wt_o   = (_Float16*)alloc(256 * 256 * 2);
  _Float16* Wt_th1 = (_Float16*)alloc(256 * 256 * 2);
  _Float16* Wt_th2 = (_Float16*)alloc(256 * 256 * 2);   // N padded 240->256
  _Float16* Wt_te1 = (_Float16*)alloc(64 * 256 * 2);    // K padded 240->256
  _Float16* Wt_te2 = (_Float16*)alloc(64 * 64 * 2);
  _Float16* Wt_sh1 = (_Float16*)alloc(256 * 1344 * 2);
  float*    qh     = (float*)alloc(MM * DD * 4);
  _Float16* Xf     = (_Float16*)alloc((size_t)ROWS_CTX * DD * 2);
  _Float16* ctxf   = (_Float16*)alloc((size_t)ROWS_CTX * DD * 2);
  _Float16* kf     = (_Float16*)alloc((size_t)ROWS_CTX * DD * 2);
  _Float16* vf     = (_Float16*)alloc((size_t)ROWS_CTX * DD * 2);
  _Float16* attnf  = (_Float16*)alloc((size_t)ROWS_MODE * DD * 2);
  float*    xres   = (float*)alloc((size_t)ROWS_MODE * DD * 4);
  _Float16* mff    = (_Float16*)alloc((size_t)ROWS_MODE * DD * 2);
  _Float16* h1f    = (_Float16*)alloc((size_t)ROWS_MODE * DD * 2);
  _Float16* flatf  = (_Float16*)alloc((size_t)ROWS_MODE * 256 * 2);
  _Float16* thf    = (_Float16*)alloc((size_t)ROWS_MODE * 64 * 2);
  _Float16* tef    = (_Float16*)alloc((size_t)ROWS_MODE * 64 * 2);
  _Float16* gf     = (_Float16*)alloc((size_t)ROWS_MODE * 256 * 2);

  auto cgrid = [](long n) { return (unsigned)((n + 255) / 256); };
  auto ggrid = [](int Mrows, int Ncols) {          // 32x64 tile per wave, 8 waves/block
    int nt = (Ncols + 63) >> 6;
    int waves = (Mrows >> 5) * nt;
    return (unsigned)((waves + 7) >> 3);
  };

  // 1) weight convert + transpose to f16 [Npad][Kpad]
  wconv<<<cgrid(256 * 256), 256, 0, stream>>>(ctx_W, Wt_ctx, 256, 256, 256, 256);
  wconv<<<cgrid(256 * 256), 256, 0, stream>>>(k_W,   Wt_k,   256, 256, 256, 256);
  wconv<<<cgrid(256 * 256), 256, 0, stream>>>(v_W,   Wt_v,   256, 256, 256, 256);
  wconv<<<cgrid(256 * 256), 256, 0, stream>>>(o_W,   Wt_o,   256, 256, 256, 256);
  wconv<<<cgrid(256 * 256), 256, 0, stream>>>(th1_W, Wt_th1, 256, 256, 256, 256);
  wconv<<<cgrid(256 * 256), 256, 0, stream>>>(th2_W, Wt_th2, 256, 240, 256, 256);
  wconv<<<cgrid(64 * 256),  256, 0, stream>>>(te1_W, Wt_te1, 240, 64, 256, 64);
  wconv<<<cgrid(64 * 64),   256, 0, stream>>>(te2_W, Wt_te2, 64, 64, 64, 64);
  wconv<<<cgrid(256 * 1344), 256, 0, stream>>>(sh1_W, Wt_sh1, 1344, 256, 1344, 256);

  // 2) q projection (broadcast queries -> computed once) and input packing
  qproj<<<MM, 256, 0, stream>>>(mode_q, q_W, q_b, qh);
  pack_inputs<<<cgrid((long)ROWS_CTX * DD), 256, 0, stream>>>(agent_map, agent_agent, neigh, map_feat, Xf);

  // 3) ctx / k / v projections (WMMA)
  gemm_wmma<0><<<ggrid(ROWS_CTX, 256), 256, 0, stream>>>(Xf, 256, Wt_ctx, 256, ctx_b,
      ROWS_CTX, 256, 256, ctxf, 256, nullptr, nullptr);
  gemm_wmma<0><<<ggrid(ROWS_CTX, 256), 256, 0, stream>>>(ctxf, 256, Wt_k, 256, k_b,
      ROWS_CTX, 256, 256, kf, 256, nullptr, nullptr);
  gemm_wmma<0><<<ggrid(ROWS_CTX, 256), 256, 0, stream>>>(ctxf, 256, Wt_v, 256, v_b,
      ROWS_CTX, 256, 256, vf, 256, nullptr, nullptr);

  // 4) skinny sigmoid cross-attention (VALU)
  attn_kernel<<<cgrid(BA * MM * 4), 256, 0, stream>>>(qh, kf, vf, attnf);

  // 5) output projection + residual (mode_queries) -> LayerNorm
  gemm_wmma<2><<<ggrid(ROWS_MODE, 256), 256, 0, stream>>>(attnf, 256, Wt_o, 256, o_b,
      ROWS_MODE, 256, 256, nullptr, 0, xres, mode_q);
  ln_rows<<<(ROWS_MODE + 7) / 8, 256, 0, stream>>>(xres, ln_g, ln_b, mff);

  // 6) trajectory head: th1 (ELU), th2 (-> trajs in d_out + flat f16 copy)
  gemm_wmma<1><<<ggrid(ROWS_MODE, 256), 256, 0, stream>>>(mff, 256, Wt_th1, 256, th1_b,
      ROWS_MODE, 256, 256, h1f, 256, nullptr, nullptr);
  gemm_wmma<3><<<ggrid(ROWS_MODE, 240), 256, 0, stream>>>(h1f, 256, Wt_th2, 256, th2_b,
      ROWS_MODE, 240, 256, flatf, 256, trajs_out, cur_state);

  // 7) trajectory embedding: te1 (ELU, K padded to 256), te2
  gemm_wmma<1><<<ggrid(ROWS_MODE, 64), 256, 0, stream>>>(flatf, 256, Wt_te1, 256, te1_b,
      ROWS_MODE, 64, 256, thf, 64, nullptr, nullptr);
  gemm_wmma<0><<<ggrid(ROWS_MODE, 64), 256, 0, stream>>>(thf, 64, Wt_te2, 64, te2_b,
      ROWS_MODE, 64, 64, tef, 64, nullptr, nullptr);

  // 8) score head: sh1 over virtual concat (K=1344), then sh2 reduction
  gemm_sh1<<<ggrid(ROWS_MODE, 256), 256, 0, stream>>>(mff, tef, ctxf, Wt_sh1, sh1_b, gf);
  score_out<<<(ROWS_MODE + 7) / 8, 256, 0, stream>>>(gf, sh2_W, sh2_b, scores_ptr);
}
